// QKVParallelLinear_79001628442668
// MI455X (gfx1250) — matrix operand order repaired and checked by reading, not compiled, not run on hardware
//
#include <hip/hip_runtime.h>

// y = x @ W^T ; the NoisePool mask r cancels exactly:
//   (x - r) W^T + W r = x W^T - (W r) + (W r) = x W^T
// GEMM: M = 8192 (B*S), N = 6144 (OUT), K = 4096 (H), f32 in/out,
// computed via V_WMMA_F32_16X16X32_BF16 with f32 accumulation.
// AI ~990 FLOP/byte -> compute bound; bf16 WMMA is the fast matrix pipe.

typedef __attribute__((ext_vector_type(16))) __bf16 v16bf;
typedef __attribute__((ext_vector_type(8)))  float  v8f;

#define BM 128
#define BN 128
#define BK 32
#define LDS_STRIDE 40   // bf16 units per row = 80 bytes: bank-conflict-free, 16B aligned

union B16x8 { __bf16 h[8]; uint4 q; };

__global__ __launch_bounds__(256)
void qkv_gemm_bf16_wmma(const float* __restrict__ x,
                        const float* __restrict__ w,
                        float* __restrict__ out,
                        int M, int N, int K)
{
    __shared__ __align__(16) __bf16 ldsX[2][BM * LDS_STRIDE];
    __shared__ __align__(16) __bf16 ldsW[2][BN * LDS_STRIDE];

    const int tid  = threadIdx.x;
    const int lane = tid & 31;
    const int wave = tid >> 5;     // 0..7
    const int wm   = wave & 1;     // 2 wave rows  -> 64 M-rows each
    const int wn   = wave >> 1;    // 4 wave cols  -> 32 N-cols each
    const int lr   = lane & 15;
    const int hi   = lane >> 4;
    const int akoff = hi * 8;      // A-fragment: K chunks {akoff..akoff+7, 16+akoff..}
    const int bkoff = hi * 8;      // B-fragment: K chunks {bkoff..bkoff+7, 16+bkoff..}

    const int m0 = blockIdx.y * BM;
    const int n0 = blockIdx.x * BN;

    // Each thread stages 16 f32 of x-tile and 16 f32 of w-tile per K-step.
    const int ldRow = tid >> 1;    // 0..127
    const int ldSeg = tid & 1;     // two 16-float segments per 32-wide row

    const float* gx = x + (size_t)(m0 + ldRow) * K + ldSeg * 16;
    const float* gw = w + (size_t)(n0 + ldRow) * K + ldSeg * 16;

    v8f acc[4][2];
#pragma unroll
    for (int mt = 0; mt < 4; ++mt)
#pragma unroll
        for (int nt = 0; nt < 2; ++nt)
            acc[mt][nt] = v8f{};

    float4 vx[4], vw[4];

    // ---- prologue: fetch + stage K-tile 0 into buffer 0 ----
    {
        const float4* px = reinterpret_cast<const float4*>(gx);
        const float4* pw = reinterpret_cast<const float4*>(gw);
#pragma unroll
        for (int i = 0; i < 4; ++i) { vx[i] = px[i]; vw[i] = pw[i]; }

#pragma unroll
        for (int half = 0; half < 2; ++half) {
            B16x8 ux, uw;
#pragma unroll
            for (int i = 0; i < 2; ++i) {
                float4 f = vx[half * 2 + i];
                ux.h[i*4+0] = (__bf16)f.x; ux.h[i*4+1] = (__bf16)f.y;
                ux.h[i*4+2] = (__bf16)f.z; ux.h[i*4+3] = (__bf16)f.w;
                float4 g = vw[half * 2 + i];
                uw.h[i*4+0] = (__bf16)g.x; uw.h[i*4+1] = (__bf16)g.y;
                uw.h[i*4+2] = (__bf16)g.z; uw.h[i*4+3] = (__bf16)g.w;
            }
            *reinterpret_cast<uint4*>(&ldsX[0][ldRow*LDS_STRIDE + ldSeg*16 + half*8]) = ux.q;
            *reinterpret_cast<uint4*>(&ldsW[0][ldRow*LDS_STRIDE + ldSeg*16 + half*8]) = uw.q;
        }
        __syncthreads();
    }

    const int KT = K / BK;   // 128 K-steps
    for (int kt = 0; kt < KT; ++kt) {
        const int cur  = kt & 1;
        const int nxt  = cur ^ 1;
        const bool more = (kt + 1) < KT;

        // Issue next tile's global loads early so VMEM hides under WMMA.
        if (more) {
            const float4* px = reinterpret_cast<const float4*>(gx + (size_t)(kt + 1) * BK);
            const float4* pw = reinterpret_cast<const float4*>(gw + (size_t)(kt + 1) * BK);
#pragma unroll
            for (int i = 0; i < 4; ++i) { vx[i] = px[i]; vw[i] = pw[i]; }
        }

        // ---- A fragments: row M = lane%16, K chunks {akoff..+7} and {16+akoff..+7} ----
        v16bf afrag[4];
#pragma unroll
        for (int mt = 0; mt < 4; ++mt) {
            const __bf16* p = &ldsX[cur][(wm*64 + mt*16 + lr) * LDS_STRIDE + akoff];
            uint4* d = reinterpret_cast<uint4*>(&afrag[mt]);
            d[0] = *reinterpret_cast<const uint4*>(p);       // K = akoff .. akoff+7
            d[1] = *reinterpret_cast<const uint4*>(p + 16);  // K = 16+akoff .. 16+akoff+7
        }
        // ---- B fragments: col N = lane%16, K chunks {bkoff..+7} and {16+bkoff..+7} ----
        v16bf bfrag[2];
#pragma unroll
        for (int nt = 0; nt < 2; ++nt) {
            const __bf16* p = &ldsW[cur][(wn*32 + nt*16 + lr) * LDS_STRIDE + bkoff];
            uint4* d = reinterpret_cast<uint4*>(&bfrag[nt]);
            d[0] = *reinterpret_cast<const uint4*>(p);       // K = bkoff .. bkoff+7
            d[1] = *reinterpret_cast<const uint4*>(p + 16);  // K = 16+bkoff .. 16+bkoff+7
        }

        // ---- 8 WMMAs: 64x32 wave tile, K += 32 ----
#pragma unroll
        for (int mt = 0; mt < 4; ++mt)
#pragma unroll
            for (int nt = 0; nt < 2; ++nt)
                acc[mt][nt] = __builtin_amdgcn_wmma_f32_16x16x32_bf16(
                    false, afrag[mt], false, bfrag[nt],
                    (short)0, acc[mt][nt], false, false);

        // ---- convert + stage next tile into the other buffer ----
        if (more) {
#pragma unroll
            for (int half = 0; half < 2; ++half) {
                B16x8 ux, uw;
#pragma unroll
                for (int i = 0; i < 2; ++i) {
                    float4 f = vx[half * 2 + i];
                    ux.h[i*4+0] = (__bf16)f.x; ux.h[i*4+1] = (__bf16)f.y;
                    ux.h[i*4+2] = (__bf16)f.z; ux.h[i*4+3] = (__bf16)f.w;
                    float4 g = vw[half * 2 + i];
                    uw.h[i*4+0] = (__bf16)g.x; uw.h[i*4+1] = (__bf16)g.y;
                    uw.h[i*4+2] = (__bf16)g.z; uw.h[i*4+3] = (__bf16)g.w;
                }
                *reinterpret_cast<uint4*>(&ldsX[nxt][ldRow*LDS_STRIDE + ldSeg*16 + half*8]) = ux.q;
                *reinterpret_cast<uint4*>(&ldsW[nxt][ldRow*LDS_STRIDE + ldSeg*16 + half*8]) = uw.q;
            }
        }
        __syncthreads();
    }

    // ---- epilogue: C layout -> global f32 ----
    // acc element v: row = mt*16 + hi*8 + v, col = wn*32 + nt*16 + lr
#pragma unroll
    for (int mt = 0; mt < 4; ++mt) {
        const int rbase = m0 + wm*64 + mt*16 + hi*8;
#pragma unroll
        for (int nt = 0; nt < 2; ++nt) {
            const int c = n0 + wn*32 + nt*16 + lr;
#pragma unroll
            for (int v = 0; v < 8; ++v)
                out[(size_t)(rbase + v) * N + c] = acc[mt][nt][v];
        }
    }
}

extern "C" void kernel_launch(void* const* d_in, const int* in_sizes, int n_in,
                              void* d_out, int out_size, void* d_ws, size_t ws_size,
                              hipStream_t stream) {
    const float* x = (const float*)d_in[0];   // [4,2048,4096] f32
    const float* w = (const float*)d_in[1];   // [6144,4096]  f32
    // d_in[2] = r: cancels algebraically, unused.
    float* out = (float*)d_out;               // [4,2048,6144] f32

    const int M = 4 * 2048;   // 8192
    const int N = 6144;
    const int K = 4096;

    dim3 grid(N / BN, M / BM);   // 48 x 64 blocks
    dim3 block(256);             // 8 wave32 per block
    qkv_gemm_bf16_wmma<<<grid, block, 0, stream>>>(x, w, out, M, N, K);
}
